// SentenceAttentionalDecoder_44375602102459
// MI455X (gfx1250) — matrix operand order repaired and checked by reading, not compiled, not run on hardware
//
#include <hip/hip_runtime.h>
#include <hip/hip_bf16.h>
#include <stdint.h>

// Problem constants (from reference)
#define B_     32
#define L_     196
#define DL_    1024
#define OUT_   512
#define G4_    2048      // 4*OUT
#define VOC_   3266
#define VOCP_  3328      // 26*128 padded
#define T_     256       // NS*NW
#define NWG_   32        // persistent WGs in recurrent kernel
#define SLICE_ 16        // OUT channels per WG
#define COLS_  64        // 4 gates * SLICE columns per WG

typedef __bf16          bf16_t;
typedef __bf16          v16bf   __attribute__((ext_vector_type(16)));
typedef float           v8f     __attribute__((ext_vector_type(8)));
typedef unsigned short  ushort8 __attribute__((ext_vector_type(8)));
typedef unsigned int    u32x4   __attribute__((ext_vector_type(4)));
typedef int             i32x4   __attribute__((ext_vector_type(4)));
typedef int             i32x8   __attribute__((ext_vector_type(8)));

__device__ __forceinline__ unsigned short f2bf(float f) {
  unsigned u = __builtin_bit_cast(unsigned, f);
  unsigned r = u + 0x7FFFu + ((u >> 16) & 1u);   // RNE
  return (unsigned short)(r >> 16);
}
__device__ __forceinline__ float sigmf(float x) { return 1.0f / (1.0f + __expf(-x)); }

// ---------------- WMMA bf16 16x16x32 fragment helpers (wave32) ----------------
// A (16x32 MxK): lane m=lane&15; lanes>=16 hold K+8; vgpr j<4 -> K 2j..; j>=4 -> K 16+...
__device__ __forceinline__ v16bf load_frag_a(const unsigned short* base, int ld, int row0, int k0) {
  int lane = threadIdx.x & 31;
  int m    = lane & 15;
  int kb   = (lane & 16) ? 8 : 0;
  const unsigned short* p = base + (size_t)(row0 + m) * ld + (k0 + kb);
  ushort8 lo = *(const ushort8*)(p);        // K = kb .. kb+7
  ushort8 hi = *(const ushort8*)(p + 16);   // K = kb+16 .. kb+23
  v16bf r;
#pragma unroll
  for (int j = 0; j < 8; ++j) {
    r[j]     = __builtin_bit_cast(bf16_t, (unsigned short)lo[j]);
    r[j + 8] = __builtin_bit_cast(bf16_t, (unsigned short)hi[j]);
  }
  return r;
}
// B (32x16 KxN) from B^T stored row-major bt[n][k]: lane n=lane&15; lanes>=16 hold K+8; j>=8 -> K 16+...
__device__ __forceinline__ v16bf load_frag_b(const unsigned short* bt, int ld, int n0, int k0) {
  int lane = threadIdx.x & 31;
  int n    = lane & 15;
  int kb   = (lane & 16) ? 8 : 0;
  const unsigned short* p = bt + (size_t)(n0 + n) * ld + (k0 + kb);
  ushort8 lo = *(const ushort8*)(p);
  ushort8 hi = *(const ushort8*)(p + 16);
  v16bf r;
#pragma unroll
  for (int j = 0; j < 8; ++j) {
    r[j]     = __builtin_bit_cast(bf16_t, (unsigned short)lo[j]);
    r[j + 8] = __builtin_bit_cast(bf16_t, (unsigned short)hi[j]);
  }
  return r;
}
__device__ __forceinline__ v8f wmma_bf16(v16bf a, v16bf b, v8f c) {
  return __builtin_amdgcn_wmma_f32_16x16x32_bf16(false, a, false, b, (short)0, c, false, false);
}

// ---------------- Tensor Data Mover: 1-D bf16 copy global -> LDS ----------------
__device__ __forceinline__ void tdm_load_1d(unsigned lds_addr, unsigned long long gaddr,
                                            unsigned nelem /* 2-byte elems, <=65535 */) {
  u32x4 g0;
  i32x8 g1;
  i32x4 g2 = {};
  i32x4 g3 = {};
  g0[0] = 1u;                                              // count=1, user mode
  g0[1] = lds_addr;                                        // lds_addr [63:32]
  g0[2] = (unsigned)(gaddr & 0xFFFFFFFFull);               // global_addr [95:64]
  g0[3] = (unsigned)((gaddr >> 32) & 0x01FFFFFFull) | 0x80000000u;  // addr hi + type=2
  unsigned td0 = nelem, td1 = 1u, tile0 = nelem;
  unsigned long long s0 = nelem;
  g1[0] = 0x00010000;                                      // data_size=1 (2B), mask=0
  g1[1] = (int)((td0 & 0xFFFFu) << 16);                    // bar_addr=0 | td0.lo
  g1[2] = (int)((td0 >> 16) | ((td1 & 0xFFFFu) << 16));    // td0.hi | td1.lo
  g1[3] = (int)((td1 >> 16) | (tile0 << 16));              // td1.hi | tile_dim0
  g1[4] = 0;                                               // tile_dim1=0(unused), tile_dim2=0
  g1[5] = (int)(s0 & 0xFFFFFFFFull);                       // tensor_dim0_stride lo
  g1[6] = (int)((s0 >> 32) & 0xFFFFull);                   // stride0 hi | stride1 lo
  g1[7] = 0;
#if __clang_major__ >= 23
  i32x8 g4 = {};
  __builtin_amdgcn_tensor_load_to_lds(g0, g1, g2, g3, g4, 0);
#else
  __builtin_amdgcn_tensor_load_to_lds(g0, g1, g2, g3, 0);
#endif
}

// ---------------- grid-wide barrier (monotonic counter, no reset races) ----------------
__device__ __forceinline__ void grid_barrier(unsigned* bar, unsigned target) {
  __threadfence();
  __syncthreads();
  if (threadIdx.x == 0) {
    __hip_atomic_fetch_add(bar, 1u, __ATOMIC_ACQ_REL, __HIP_MEMORY_SCOPE_AGENT);
    while (__hip_atomic_load(bar, __ATOMIC_ACQUIRE, __HIP_MEMORY_SCOPE_AGENT) < target)
      __builtin_amdgcn_s_sleep(1);
  }
  __syncthreads();
}

// ================= prep kernels =================
__global__ void k_init(unsigned* bar, unsigned short* hbuf, unsigned short* h1buf) {
  int i = blockIdx.x * blockDim.x + threadIdx.x;
  if (i == 0) *bar = 0u;
  for (int j = i; j < B_ * OUT_; j += gridDim.x * blockDim.x) { hbuf[j] = 0; h1buf[j] = 0; }
}

// x[b,d] = sum_l local_features[b,d,l]  (attention collapses to all-ones weights)
// one wave per row: coalesced lane-parallel reduction + wave32 shuffle tree
__global__ void k_redx(const float* lf, unsigned short* xb) {
  int row  = blockIdx.x * (blockDim.x >> 5) + (threadIdx.x >> 5);
  int lane = threadIdx.x & 31;
  if (row >= B_ * DL_) return;
  const float* p = lf + (size_t)row * L_;
  float s = 0.f;
  for (int j = lane; j < L_; j += 32) s += p[j];
#pragma unroll
  for (int o = 16; o > 0; o >>= 1) s += __shfl_down(s, o, 32);
  if (lane == 0) xb[row] = f2bf(s);
}

__global__ void k_bias2(const float* bih2, const float* bhh2, float* bias2) {
  int i = blockIdx.x * blockDim.x + threadIdx.x;
  if (i < G4_) bias2[i] = bih2[i] + bhh2[i];
}

__global__ void k_conv_wih1(const float* s, unsigned short* d) {
  for (size_t i = blockIdx.x * (size_t)blockDim.x + threadIdx.x; i < (size_t)G4_ * DL_;
       i += (size_t)gridDim.x * blockDim.x)
    d[i] = f2bf(s[i]);
}

__global__ void k_conv_wf(const float* s, unsigned short* d) {
  for (size_t i = blockIdx.x * (size_t)blockDim.x + threadIdx.x; i < (size_t)VOCP_ * OUT_;
       i += (size_t)gridDim.x * blockDim.x) {
    size_t n = i / OUT_, k = i % OUT_;
    d[i] = (n < VOC_) ? f2bf(s[n * OUT_ + k]) : (unsigned short)0;
  }
}

// blob[wg][m][col][k]: per-WG column slices of W_hh1(m=0), W_ih2(m=1), W_hh2(m=2)
__global__ void k_conv_blob(const float* whh1, const float* wih2, const float* whh2,
                            unsigned short* d) {
  for (size_t i = blockIdx.x * (size_t)blockDim.x + threadIdx.x; i < (size_t)NWG_ * 3 * COLS_ * OUT_;
       i += (size_t)gridDim.x * blockDim.x) {
    int k = (int)(i & (OUT_ - 1));
    size_t r1 = i >> 9;
    int col = (int)(r1 & (COLS_ - 1));
    size_t r2 = r1 >> 6;
    int m  = (int)(r2 % 3);
    int wg = (int)(r2 / 3);
    int g = col >> 4, c = col & 15;
    int row = g * OUT_ + wg * SLICE_ + c;
    const float* src = (m == 0) ? whh1 : (m == 1) ? wih2 : whh2;
    d[i] = f2bf(src[(size_t)row * OUT_ + k]);
  }
}

// pre1[32,2048] = x_bf16 @ W_ih1^T + b_ih1 + b_hh1   (constant across all timesteps)
__global__ __launch_bounds__(256) void k_pre1(const unsigned short* xb, const unsigned short* wih1b,
                                              const float* bih1, const float* bhh1, float* pre1) {
  int wave = threadIdx.x >> 5, lane = threadIdx.x & 31;
  int mt = wave & 1, nt = wave >> 1;
  int n0 = blockIdx.x * 64 + nt * 16;
  v8f acc = {};
#pragma unroll 4
  for (int k = 0; k < DL_; k += 32) {
    v16bf a = load_frag_a(xb, DL_, mt * 16, k);
    v16bf b = load_frag_b(wih1b, DL_, n0, k);
    acc = wmma_bf16(a, b, acc);
  }
  int n = lane & 15, mb = (lane & 16) ? 8 : 0;
  int col = n0 + n;
  float bias = bih1[col] + bhh1[col];
#pragma unroll
  for (int r = 0; r < 8; ++r) {
    int row = mt * 16 + mb + r;
    pre1[(size_t)row * G4_ + col] = acc[r] + bias;
  }
}

// ================= sequential dual-LSTM, 32 persistent WGs, 2 grid syncs/step =================
// Probe the documented 320KB/WGP LDS: keep ALL recurrent weight slices (192KB) LDS-resident
// for the whole 256-step chain; zero per-step L2 weight traffic. Total static LDS ~276KB.
__global__ __launch_bounds__(256) void k_rec(const unsigned short* blob, const float* pre1,
                                             const float* bias2, unsigned short* hbuf,
                                             unsigned short* h1buf, unsigned short* h1all,
                                             unsigned* bar) {
  __shared__ unsigned short wsl[3][COLS_][OUT_]; // W_hh1/W_ih2/W_hh2 slices, 192KB, resident
  __shared__ unsigned short hA[B_][OUT_];        // staged h   (32KB)
  __shared__ unsigned short h1A[B_][OUT_];       // staged h1  (32KB)
  __shared__ float gat[4][B_][SLICE_];           // i/f/g/o exchange (8KB)
  __shared__ float cst[2][B_][SLICE_];           // cell states c, c1 (4KB)
  __shared__ float preS[B_][COLS_];              // pre1 slice (8KB)
  __shared__ float b2S[COLS_];

  const int wg = blockIdx.x;
  const int tid = threadIdx.x;
  const int wave = tid >> 5, lane = tid & 31;
  const int mt = wave & 1, nt = wave >> 1;

  if (wave == 0) {  // TDM: all three weight slices -> LDS (6 x 16384 bf16 = 192KB)
    const unsigned short* src = blob + (size_t)wg * 3 * COLS_ * OUT_;
    unsigned lds0 = (unsigned)(uintptr_t)&wsl[0][0][0];
#pragma unroll
    for (int part = 0; part < 6; ++part)
      tdm_load_1d(lds0 + (unsigned)part * 32768u,
                  (unsigned long long)(uintptr_t)(src + (size_t)part * 16384), 16384u);
    __builtin_amdgcn_s_wait_tensorcnt(0);
  }
  for (int i = tid; i < B_ * COLS_; i += 256) {
    int b = i >> 6, c = i & 63;
    int rr = (c >> 4) * OUT_ + wg * SLICE_ + (c & 15);
    preS[b][c] = pre1[(size_t)b * G4_ + rr];
    if (b == 0) b2S[c] = bias2[rr];
  }
  for (int i = tid; i < 2 * B_ * SLICE_; i += 256) ((float*)cst)[i] = 0.f;
  __syncthreads();

  unsigned ev = 0;
  for (int t = 0; t < T_; ++t) {
    const int p = t & 1, pn = p ^ 1;
    if (wave == 0) {  // stage h_prev, h1_prev
      tdm_load_1d((unsigned)(uintptr_t)&hA[0][0],
                  (unsigned long long)(uintptr_t)(hbuf + (size_t)p * B_ * OUT_), B_ * OUT_);
      tdm_load_1d((unsigned)(uintptr_t)&h1A[0][0],
                  (unsigned long long)(uintptr_t)(h1buf + (size_t)p * B_ * OUT_), B_ * OUT_);
      __builtin_amdgcn_s_wait_tensorcnt(0);
    }
    __syncthreads();

    { // ---- cell1: gates = pre1_slice + h_prev @ W_hh1_slice ----
      v8f acc = {};
#pragma unroll
      for (int k = 0; k < OUT_; k += 32) {
        v16bf a = load_frag_a(&hA[0][0], OUT_, mt * 16, k);
        v16bf b = load_frag_b(&wsl[0][0][0], OUT_, nt * 16, k);
        acc = wmma_bf16(a, b, acc);
      }
      int n = lane & 15, mb = (lane & 16) ? 8 : 0;
#pragma unroll
      for (int r = 0; r < 8; ++r) {
        int b = mt * 16 + mb + r;
        gat[nt][b][n] = acc[r] + preS[b][nt * 16 + n];
      }
    }
    __syncthreads();
    for (int i = tid; i < B_ * SLICE_; i += 256) {
      int b = i >> 4, ch = i & 15;
      float cn = sigmf(gat[1][b][ch]) * cst[0][b][ch] + sigmf(gat[0][b][ch]) * tanhf(gat[2][b][ch]);
      cst[0][b][ch] = cn;
      float h = sigmf(gat[3][b][ch]) * tanhf(cn);
      hbuf[(size_t)pn * B_ * OUT_ + (size_t)b * OUT_ + wg * SLICE_ + ch] = f2bf(h);
    }
    grid_barrier(bar, (++ev) * NWG_);

    if (wave == 0) {  // stage full h_new
      tdm_load_1d((unsigned)(uintptr_t)&hA[0][0],
                  (unsigned long long)(uintptr_t)(hbuf + (size_t)pn * B_ * OUT_), B_ * OUT_);
      __builtin_amdgcn_s_wait_tensorcnt(0);
    }
    __syncthreads();

    { // ---- cell2: gates = bias2 + h_new @ W_ih2_s + h1_prev @ W_hh2_s (all LDS) ----
      v8f acc = {};
#pragma unroll
      for (int k = 0; k < OUT_; k += 32) {
        v16bf a = load_frag_a(&hA[0][0], OUT_, mt * 16, k);
        v16bf b = load_frag_b(&wsl[1][0][0], OUT_, nt * 16, k);
        acc = wmma_bf16(a, b, acc);
      }
#pragma unroll
      for (int k = 0; k < OUT_; k += 32) {
        v16bf a = load_frag_a(&h1A[0][0], OUT_, mt * 16, k);
        v16bf b = load_frag_b(&wsl[2][0][0], OUT_, nt * 16, k);
        acc = wmma_bf16(a, b, acc);
      }
      int n = lane & 15, mb = (lane & 16) ? 8 : 0;
#pragma unroll
      for (int r = 0; r < 8; ++r) {
        int b = mt * 16 + mb + r;
        gat[nt][b][n] = acc[r] + b2S[nt * 16 + n];
      }
    }
    __syncthreads();
    for (int i = tid; i < B_ * SLICE_; i += 256) {
      int b = i >> 4, ch = i & 15;
      float cn = sigmf(gat[1][b][ch]) * cst[1][b][ch] + sigmf(gat[0][b][ch]) * tanhf(gat[2][b][ch]);
      cst[1][b][ch] = cn;
      unsigned short hb = f2bf(sigmf(gat[3][b][ch]) * tanhf(cn));
      h1buf[(size_t)pn * B_ * OUT_ + (size_t)b * OUT_ + wg * SLICE_ + ch] = hb;
      h1all[((size_t)t * B_ + b) * OUT_ + wg * SLICE_ + ch] = hb;
    }
    grid_barrier(bar, (++ev) * NWG_);
  }
}

// ================= output GEMM: words = h1_all @ W_f^T + b_f, scattered to [B,NS,NW,V] =================
__global__ __launch_bounds__(256) void k_wf(const unsigned short* h1all, const unsigned short* wfb,
                                            const float* bfv, float* out) {
  __shared__ unsigned short hA[B_][OUT_];
  const int t = blockIdx.y;
  const int wave = threadIdx.x >> 5, lane = threadIdx.x & 31;
  const int mt = wave & 1;
  if (wave == 0) {
    tdm_load_1d((unsigned)(uintptr_t)&hA[0][0],
                (unsigned long long)(uintptr_t)(h1all + (size_t)t * B_ * OUT_), B_ * OUT_);
    __builtin_amdgcn_s_wait_tensorcnt(0);
  }
  __syncthreads();
  const int n0 = blockIdx.x * 128 + (wave >> 1) * 32;
  v8f acc0 = {}, acc1 = {};
#pragma unroll 4
  for (int k = 0; k < OUT_; k += 32) {
    v16bf a  = load_frag_a(&hA[0][0], OUT_, mt * 16, k);
    v16bf b0 = load_frag_b(wfb, OUT_, n0, k);
    v16bf b1 = load_frag_b(wfb, OUT_, n0 + 16, k);
    acc0 = wmma_bf16(a, b0, acc0);
    acc1 = wmma_bf16(a, b1, acc1);
  }
  int n = lane & 15, mb = (lane & 16) ? 8 : 0;
#pragma unroll
  for (int r = 0; r < 8; ++r) {
    int b = mt * 16 + mb + r;
    size_t rowbase = ((size_t)b * T_ + t) * VOC_;   // word_list[b, t/NW, t%NW, :]
    int v0 = n0 + n, v1 = n0 + 16 + n;
    if (v0 < VOC_) out[rowbase + v0] = acc0[r] + bfv[v0];
    if (v1 < VOC_) out[rowbase + v1] = acc1[r] + bfv[v1];
  }
}

// ================= host glue =================
extern "C" void kernel_launch(void* const* d_in, const int* in_sizes, int n_in,
                              void* d_out, int out_size, void* d_ws, size_t ws_size,
                              hipStream_t stream) {
  (void)in_sizes; (void)n_in; (void)out_size; (void)ws_size;
  const float* lf   = (const float*)d_in[0];
  // d_in[1..4] (sentenc, W_s, W_v, W_a) are dead: softmax over a size-1 axis => weights == 1
  const float* Wih1 = (const float*)d_in[5];
  const float* Whh1 = (const float*)d_in[6];
  const float* bih1 = (const float*)d_in[7];
  const float* bhh1 = (const float*)d_in[8];
  const float* Wih2 = (const float*)d_in[9];
  const float* Whh2 = (const float*)d_in[10];
  const float* bih2 = (const float*)d_in[11];
  const float* bhh2 = (const float*)d_in[12];
  const float* Wf   = (const float*)d_in[13];
  const float* bfv  = (const float*)d_in[14];
  float* out = (float*)d_out;

  char* w = (char*)d_ws;
  size_t off = 0;
  auto take = [&](size_t bytes) -> void* {
    void* p = w + off;
    off = (off + bytes + 255) & ~(size_t)255;
    return p;
  };
  unsigned*       bar    = (unsigned*)take(256);
  unsigned short* xb     = (unsigned short*)take((size_t)B_ * DL_ * 2);
  unsigned short* wih1b  = (unsigned short*)take((size_t)G4_ * DL_ * 2);
  unsigned short* wfb    = (unsigned short*)take((size_t)VOCP_ * OUT_ * 2);
  unsigned short* blob   = (unsigned short*)take((size_t)NWG_ * 3 * COLS_ * OUT_ * 2);
  float*          bias2  = (float*)take((size_t)G4_ * 4);
  float*          pre1   = (float*)take((size_t)B_ * G4_ * 4);
  unsigned short* hbuf   = (unsigned short*)take((size_t)2 * B_ * OUT_ * 2);
  unsigned short* h1buf  = (unsigned short*)take((size_t)2 * B_ * OUT_ * 2);
  unsigned short* h1all  = (unsigned short*)take((size_t)T_ * B_ * OUT_ * 2);

  k_init<<<dim3(64), dim3(256), 0, stream>>>(bar, hbuf, h1buf);
  k_redx<<<dim3((B_ * DL_ + 7) / 8), dim3(256), 0, stream>>>(lf, xb);
  k_bias2<<<dim3(8), dim3(256), 0, stream>>>(bih2, bhh2, bias2);
  k_conv_wih1<<<dim3(1024), dim3(256), 0, stream>>>(Wih1, wih1b);
  k_conv_wf<<<dim3(1024), dim3(256), 0, stream>>>(Wf, wfb);
  k_conv_blob<<<dim3(1024), dim3(256), 0, stream>>>(Whh1, Wih2, Whh2, blob);
  k_pre1<<<dim3(G4_ / 64), dim3(256), 0, stream>>>(xb, wih1b, bih1, bhh1, pre1);
  k_rec<<<dim3(NWG_), dim3(256), 0, stream>>>(blob, pre1, bias2, hbuf, h1buf, h1all, bar);
  k_wf<<<dim3(VOCP_ / 128, T_), dim3(256), 0, stream>>>(h1all, wfb, bfv, out);
}
